// ReferenceLocalTransformerBlock2d_46110768890401
// MI455X (gfx1250) — matrix operand order repaired and checked by reading, not compiled, not run on hardware
//
#include <hip/hip_runtime.h>
#include <math.h>

// ---------------------------------------------------------------------------
// Shapes (fixed by the reference): B=1, C=256, H=W=56, N=3136, heads=8, d=32
// ---------------------------------------------------------------------------
#define NPIX   3136
#define CCH    256
#define HEADS  8
#define HDIM   32
#define HW     56

typedef __attribute__((ext_vector_type(16))) __bf16 v16bf;
typedef __attribute__((ext_vector_type(8)))  __bf16 v8bf;
typedef __attribute__((ext_vector_type(8)))  float  v8f;

__device__ __forceinline__ unsigned short f2bf(float f) {
    unsigned int u = __float_as_uint(f);
    unsigned int r = u + 0x7FFFu + ((u >> 16) & 1u);   // round-to-nearest-even
    return (unsigned short)(r >> 16);
}

// A-fragment: 16-bit A 16x32 fragment, same k order as B. Lane holds row (lane&15);
// lane-half selects K = {half*16..+15} (contiguous) -> two 16B loads.
__device__ __forceinline__ v16bf load_afrag(const __bf16* __restrict__ Arow,
                                            int k0, int half) {
    v8bf lo = *(const v8bf*)(Arow + k0 + half * 16);
    v8bf hi = *(const v8bf*)(Arow + k0 + half * 16 + 8);
    return __builtin_shufflevector(lo, hi,
               0,1,2,3,4,5,6,7,8,9,10,11,12,13,14,15);
}

// ------------------------- fp32 -> bf16 weight prep -------------------------
__global__ void k_tobf16(const float* __restrict__ src,
                         unsigned short* __restrict__ dst, int n) {
    int i = blockIdx.x * 256 + threadIdx.x;
    if (i < n) dst[i] = f2bf(src[i]);
}

// ---------------- channel LayerNorm:  x[C,N] -> t[N,C] (bf16) ---------------
__global__ void k_channel_ln(const float* __restrict__ x,
                             const float* __restrict__ w,
                             const float* __restrict__ b,
                             unsigned short* __restrict__ out) {
    int wid  = threadIdx.x >> 5;
    int lane = threadIdx.x & 31;
    int n    = blockIdx.x * 8 + wid;            // 392 blocks * 8 waves = 3136
    float v[8]; float s = 0.f, ss = 0.f;
#pragma unroll
    for (int t = 0; t < 8; ++t) {
        int c   = lane + 32 * t;
        float f = x[(size_t)c * NPIX + n];
        v[t] = f; s += f; ss += f * f;
    }
#pragma unroll
    for (int off = 16; off; off >>= 1) {
        s  += __shfl_xor(s,  off, 32);
        ss += __shfl_xor(ss, off, 32);
    }
    float mean = s * (1.f / 256.f);
    float var  = ss * (1.f / 256.f) - mean * mean;
    float rstd = rsqrtf(var + 1e-5f);
#pragma unroll
    for (int t = 0; t < 8; ++t) {
        int c = lane + 32 * t;
        float xh = (v[t] - mean) * rstd * w[c] + b[c];
        out[(size_t)n * CCH + c] = f2bf(xh);
    }
}

// ------------------------------ WMMA GEMM ----------------------------------
// D[N,M] = A[N,K](bf16) * W[M,K](bf16)^T + bias.
// 32x32 output per wave = 2x2 WMMA register block (4 independent acc chains).
// MODE 0: store f32 [N,M]          (QKV)
// MODE 1: GELU(exact erf) -> bf16  (fc1)
// MODE 2: +bias, +resid[C,N], transpose-store f32 [C,N]  (proj / fc2)
template <int MODE>
__device__ __forceinline__ void store_tile(v8f acc, int rbase, int cbase,
                                           int lane, int M,
                                           const float* __restrict__ bias,
                                           float* __restrict__ outF,
                                           unsigned short* __restrict__ outBf,
                                           const float* __restrict__ resid) {
    int half = lane >> 4;
    int col  = cbase + (lane & 15);
    float bv = bias[col];
#pragma unroll
    for (int j = 0; j < 8; ++j) {
        int r   = rbase + j + half * 8;         // C/D layout: M = j + 8*half
        float v = acc[j] + bv;
        if (MODE == 0) {
            outF[(size_t)r * M + col] = v;
        } else if (MODE == 1) {
            float g = 0.5f * v * (1.f + erff(v * 0.70710678118654752f));
            outBf[(size_t)r * M + col] = f2bf(g);
        } else {
            size_t idx = (size_t)col * NPIX + r;  // transpose to [C,N]
            outF[idx] = v + resid[idx];
        }
    }
}

template <int MODE>
__global__ void k_gemm_wmma(const __bf16* __restrict__ A,
                            const __bf16* __restrict__ W,
                            const float* __restrict__ bias,
                            int M, int K, int totalTiles,
                            float* __restrict__ outF,
                            unsigned short* __restrict__ outBf,
                            const float* __restrict__ resid) {
    int wid  = threadIdx.x >> 5;
    int lane = threadIdx.x & 31;
    int tile = blockIdx.x * 8 + wid;
    if (tile >= totalTiles) return;             // wave-uniform exit

    const int rowTiles = NPIX / 32;             // 98 (32-row super-tiles)
    int rt = tile % rowTiles;
    int ct = tile / rowTiles;

    int half  = lane >> 4;
    int rowA0 = rt * 32 + (lane & 15);          // A rows for the two sub-tiles
    int colB0 = ct * 32 + (lane & 15);          // output features (B columns)

    const __bf16* Arow0 = A + (size_t)rowA0 * K;
    const __bf16* Arow1 = Arow0 + (size_t)16 * K;
    const __bf16* Wrow0 = W + (size_t)colB0 * K + half * 16;  // 16 contig K
    const __bf16* Wrow1 = Wrow0 + (size_t)16 * K;

    v8f acc00 = {}, acc01 = {}, acc10 = {}, acc11 = {};
    for (int k0 = 0; k0 < K; k0 += 32) {
        v16bf a0 = load_afrag(Arow0, k0, half);
        v16bf a1 = load_afrag(Arow1, k0, half);
        v16bf b0 = *(const v16bf*)(Wrow0 + k0);
        v16bf b1 = *(const v16bf*)(Wrow1 + k0);
        acc00 = __builtin_amdgcn_wmma_f32_16x16x32_bf16(
                    false, a0, false, b0, (short)0, acc00, false, false);
        acc01 = __builtin_amdgcn_wmma_f32_16x16x32_bf16(
                    false, a0, false, b1, (short)0, acc01, false, false);
        acc10 = __builtin_amdgcn_wmma_f32_16x16x32_bf16(
                    false, a1, false, b0, (short)0, acc10, false, false);
        acc11 = __builtin_amdgcn_wmma_f32_16x16x32_bf16(
                    false, a1, false, b1, (short)0, acc11, false, false);
    }

    store_tile<MODE>(acc00, rt * 32,      ct * 32,      lane, M, bias, outF, outBf, resid);
    store_tile<MODE>(acc01, rt * 32,      ct * 32 + 16, lane, M, bias, outF, outBf, resid);
    store_tile<MODE>(acc10, rt * 32 + 16, ct * 32,      lane, M, bias, outF, outBf, resid);
    store_tile<MODE>(acc11, rt * 32 + 16, ct * 32 + 16, lane, M, bias, outF, outBf, resid);
}

// --------------------- local windowed attention (w=7) ----------------------
// One wave per (pixel, head); lane = head-dim. QKV layout: [N, 3*256] with
// q at +h*32, k at +256+h*32, v at +512+h*32.
__global__ void k_local_attn(const float* __restrict__ QKV,
                             unsigned short* __restrict__ outA) {
    __shared__ float sc[8][52];
    int wid  = threadIdx.x >> 5;
    int lane = threadIdx.x & 31;
    int idx  = blockIdx.x * 8 + wid;            // 3136 blocks * 8 = N*heads
    int n = idx >> 3;
    int h = idx & 7;
    int i = n / HW, j = n % HW;
    const float scale = 0.17677669529663687f;   // 1/sqrt(32)

    float qv = QKV[(size_t)n * 768 + h * 32 + lane];
    float mx = -1e30f;
    int cnt = 0;
    for (int di = -3; di <= 3; ++di) {
        int ii = i + di; if ((unsigned)ii >= (unsigned)HW) continue;
        for (int dj = -3; dj <= 3; ++dj) {
            int jj = j + dj; if ((unsigned)jj >= (unsigned)HW) continue;
            int m = ii * HW + jj;
            float p = qv * QKV[(size_t)m * 768 + 256 + h * 32 + lane];
#pragma unroll
            for (int off = 16; off; off >>= 1) p += __shfl_xor(p, off, 32);
            p *= scale;                         // every lane has the score
            if (lane == 0) sc[wid][cnt] = p;
            mx = fmaxf(mx, p);
            ++cnt;
        }
    }
    __syncthreads();
    float denom = 0.f, acc = 0.f;
    cnt = 0;
    for (int di = -3; di <= 3; ++di) {
        int ii = i + di; if ((unsigned)ii >= (unsigned)HW) continue;
        for (int dj = -3; dj <= 3; ++dj) {
            int jj = j + dj; if ((unsigned)jj >= (unsigned)HW) continue;
            int m = ii * HW + jj;
            float pe = __expf(sc[wid][cnt] - mx); ++cnt;
            denom += pe;
            acc   += pe * QKV[(size_t)m * 768 + 512 + h * 32 + lane];
        }
    }
    outA[(size_t)n * CCH + h * 32 + lane] = f2bf(acc / denom);
}

// ---------------------------------------------------------------------------
extern "C" void kernel_launch(void* const* d_in, const int* in_sizes, int n_in,
                              void* d_out, int out_size, void* d_ws, size_t ws_size,
                              hipStream_t stream) {
    (void)in_sizes; (void)n_in; (void)out_size; (void)ws_size;

    const float* x       = (const float*)d_in[0];
    const float* norm1_w = (const float*)d_in[1];
    const float* norm1_b = (const float*)d_in[2];
    const float* qkv_w   = (const float*)d_in[3];
    const float* qkv_b   = (const float*)d_in[4];
    const float* proj_w  = (const float*)d_in[5];
    const float* proj_b  = (const float*)d_in[6];
    const float* norm2_w = (const float*)d_in[7];
    const float* norm2_b = (const float*)d_in[8];
    const float* fc1_w   = (const float*)d_in[9];
    const float* fc1_b   = (const float*)d_in[10];
    const float* fc2_w   = (const float*)d_in[11];
    const float* fc2_b   = (const float*)d_in[12];
    float* out = (float*)d_out;                 // [C, N] fp32

    // ---- workspace carve-up (all offsets 256B aligned) ----
    char* ws = (char*)d_ws;
    size_t off = 0;
    auto take = [&](size_t bytes) { char* p = ws + off;
                                    off += (bytes + 255) & ~(size_t)255; return p; };
    unsigned short* qkvW  = (unsigned short*)take((size_t)768 * 256 * 2);
    unsigned short* projW = (unsigned short*)take((size_t)256 * 256 * 2);
    unsigned short* fc1W  = (unsigned short*)take((size_t)1024 * 256 * 2);
    unsigned short* fc2W  = (unsigned short*)take((size_t)1024 * 256 * 2);
    unsigned short* tAct  = (unsigned short*)take((size_t)NPIX * 256 * 2);  // LN out (reused)
    float*          QKV   = (float*)take((size_t)NPIX * 768 * 4);
    unsigned short* attnA = (unsigned short*)take((size_t)NPIX * 256 * 2);
    float*          x1    = (float*)take((size_t)NPIX * 256 * 4);           // [C,N]
    unsigned short* hBuf  = (unsigned short*)take((size_t)NPIX * 1024 * 2);

    // ---- weights -> bf16 ----
    k_tobf16<<<(768 * 256 + 255) / 256, 256, 0, stream>>>(qkv_w,  qkvW,  768 * 256);
    k_tobf16<<<(256 * 256 + 255) / 256, 256, 0, stream>>>(proj_w, projW, 256 * 256);
    k_tobf16<<<(1024 * 256 + 255) / 256, 256, 0, stream>>>(fc1_w, fc1W, 1024 * 256);
    k_tobf16<<<(1024 * 256 + 255) / 256, 256, 0, stream>>>(fc2_w, fc2W, 1024 * 256);

    const int LN_BLOCKS = NPIX / 8;             // 392

    // ---- LN1 ----
    k_channel_ln<<<LN_BLOCKS, 256, 0, stream>>>(x, norm1_w, norm1_b, tAct);

    // ---- QKV GEMM: [N,256] x [768,256]^T -> f32 [N,768] ----
    {
        int tiles = (NPIX / 32) * (768 / 32);   // 98 * 24 = 2352
        k_gemm_wmma<0><<<(tiles + 7) / 8, 256, 0, stream>>>(
            (const __bf16*)tAct, (const __bf16*)qkvW, qkv_b,
            768, 256, tiles, QKV, nullptr, nullptr);
    }

    // ---- local attention -> bf16 [N,256] ----
    k_local_attn<<<NPIX, 256, 0, stream>>>(QKV, attnA);

    // ---- proj GEMM + residual(x) -> f32 [C,N] ----
    {
        int tiles = (NPIX / 32) * (256 / 32);   // 98 * 8 = 784
        k_gemm_wmma<2><<<(tiles + 7) / 8, 256, 0, stream>>>(
            (const __bf16*)attnA, (const __bf16*)projW, proj_b,
            256, 256, tiles, x1, nullptr, x);
    }

    // ---- LN2 ----
    k_channel_ln<<<LN_BLOCKS, 256, 0, stream>>>(x1, norm2_w, norm2_b, tAct);

    // ---- fc1 GEMM + GELU -> bf16 [N,1024] ----
    {
        int tiles = (NPIX / 32) * (1024 / 32);  // 98 * 32 = 3136
        k_gemm_wmma<1><<<(tiles + 7) / 8, 256, 0, stream>>>(
            (const __bf16*)tAct, (const __bf16*)fc1W, fc1_b,
            1024, 256, tiles, nullptr, hBuf, nullptr);
    }

    // ---- fc2 GEMM + residual(x1) -> f32 [C,N] = d_out ----
    {
        int tiles = (NPIX / 32) * (256 / 32);   // 784
        k_gemm_wmma<2><<<(tiles + 7) / 8, 256, 0, stream>>>(
            (const __bf16*)hBuf, (const __bf16*)fc2W, fc2_b,
            256, 1024, tiles, out, nullptr, x1);
    }
}
